// GCNLayer_25177098289616
// MI455X (gfx1250) — matrix operand order repaired and checked by reading, not compiled, not run on hardware
//
#include <hip/hip_runtime.h>
#include <hip/hip_bf16.h>

typedef __attribute__((ext_vector_type(16))) _Float16 v16h;
typedef __attribute__((ext_vector_type(8)))  _Float16 v8h;
typedef __attribute__((ext_vector_type(8)))  float    v8f;

#define F_DIM     128
#define LDS_PITCH 136   // halves; 272B rows -> 4-bank rotation per row, no conflicts

// ---------------------------------------------------------------------------
// Kernel 1: W [128x128] f32 row-major  ->  Wt [128x128] f16 with Wt[n][k]=W[k][n]
// one block of 256 threads; 64 elements per thread, coalesced reads.
// ---------------------------------------------------------------------------
__global__ __launch_bounds__(256) void prep_w_kernel(const float* __restrict__ W,
                                                     _Float16* __restrict__ Wt) {
    int tid = threadIdx.x;
    for (int j = 0; j < 64; ++j) {
        int idx = j * 256 + tid;          // 0 .. 16383
        int k = idx >> 7;                 // row of W
        int n = idx & 127;                // col of W
        Wt[n * F_DIM + k] = (_Float16)W[idx];
    }
}

// ---------------------------------------------------------------------------
// Kernel 2: XW = X @ W via v_wmma_f32_16x16x32_f16, fp32 accumulate.
// Block = 256 threads = 8 waves; each wave: one 16-row tile x 128 cols.
// ---------------------------------------------------------------------------
__global__ __launch_bounds__(256) void gemm_wmma_kernel(const float* __restrict__ X,
                                                        const _Float16* __restrict__ Wt,
                                                        float* __restrict__ XW,
                                                        int Nrows) {
    __shared__ _Float16 lds[F_DIM * LDS_PITCH];   // 34.8 KB of the 320 KB WGP LDS

    const int tid       = threadIdx.x;
    const int lane      = tid & 31;
    const int waveInBlk = tid >> 5;

    // Stage Wt (f16, transposed) into LDS with padded pitch. 16-B vector copies.
    #pragma unroll
    for (int j = 0; j < 8; ++j) {
        int idx = (j * 256 + tid) * 8;            // half index, 16-B aligned
        int n   = idx >> 7;
        int k   = idx & 127;
        *(uint4*)&lds[n * LDS_PITCH + k] = *(const uint4*)&Wt[idx];
    }
    __syncthreads();

    const int numTiles = Nrows >> 4;
    const int tile = blockIdx.x * 8 + waveInBlk;  // wave-uniform
    if (tile >= numTiles) return;

    // A-fragment row for this lane (ISA 16-bit A 16x32 layout)
    int m = tile * 16 + (lane & 15);
    if (m >= Nrows) m = Nrows - 1;                // clamp (N is a multiple of 16 here)
    const float* xrow = X + (size_t)m * F_DIM;
    const int kbase = (lane >= 16) ? 16 : 0;      // lanes>=16 hold K=16..31
    const int koffB = (lane >= 16) ? 16 : 0;      // B: lanes>=16 hold K=16..31

    v8f acc[8] = {};

    #pragma unroll
    for (int c = 0; c < 4; ++c) {                 // K chunks of 32
        const int k0 = c * 32;

        // Build A fragment: halves 0-7 = X[m][k0+kbase .. +7],
        //                   halves 8-15 = X[m][k0+kbase+8 .. +15]
        const float* p0 = xrow + k0 + kbase;
        const float* p1 = xrow + k0 + kbase + 8;
        float4 xa = *(const float4*)(p0);
        float4 xb = *(const float4*)(p0 + 4);
        float4 xc = *(const float4*)(p1);
        float4 xd = *(const float4*)(p1 + 4);
        v16h a;
        a[0]  = (_Float16)xa.x; a[1]  = (_Float16)xa.y; a[2]  = (_Float16)xa.z; a[3]  = (_Float16)xa.w;
        a[4]  = (_Float16)xb.x; a[5]  = (_Float16)xb.y; a[6]  = (_Float16)xb.z; a[7]  = (_Float16)xb.w;
        a[8]  = (_Float16)xc.x; a[9]  = (_Float16)xc.y; a[10] = (_Float16)xc.z; a[11] = (_Float16)xc.w;
        a[12] = (_Float16)xd.x; a[13] = (_Float16)xd.y; a[14] = (_Float16)xd.z; a[15] = (_Float16)xd.w;

        #pragma unroll
        for (int t = 0; t < 8; ++t) {             // 8 N-tiles of 16 cols
            // B fragment (32x16 KxN): lane holds 16 contiguous halves of Wt row (col n)
            const _Float16* bp =
                &lds[(t * 16 + (lane & 15)) * LDS_PITCH + k0 + koffB];
            v8h blo = *(const v8h*)bp;            // ds_load_b128
            v8h bhi = *(const v8h*)(bp + 8);      // ds_load_b128
            v16h b;
            #pragma unroll
            for (int h = 0; h < 8; ++h) { b[h] = blo[h]; b[h + 8] = bhi[h]; }

            acc[t] = __builtin_amdgcn_wmma_f32_16x16x32_f16(
                false, a, false, b, (short)0, acc[t], false, false);
        }
    }

    // Store D (f32 16x16 layout: VGPR r -> row r (lanes<16) / r+8 (lanes>=16))
    const int row0 = tile * 16 + ((lane >= 16) ? 8 : 0);
    const int col  = lane & 15;
    #pragma unroll
    for (int t = 0; t < 8; ++t) {
        #pragma unroll
        for (int r = 0; r < 8; ++r) {
            int row = row0 + r;
            if (row < Nrows)
                XW[(size_t)row * F_DIM + t * 16 + col] = acc[t][r];
        }
    }
}

// ---------------------------------------------------------------------------
// Kernel 3: CSR aggregation. One wave per node: 32 lanes x float4 = one row.
// Edges staged across lanes, broadcast via __shfl (wave32). Gathers hit L2
// (XW = 51.2 MB << 192 MB global L2).
// ---------------------------------------------------------------------------
__global__ __launch_bounds__(256) void agg_kernel(const int* __restrict__ row_ptr,
                                                  const int* __restrict__ col_idx,
                                                  const float* __restrict__ values,
                                                  const float* __restrict__ XW,
                                                  float* __restrict__ out,
                                                  int N) {
    const int node = (int)((blockIdx.x * (unsigned)blockDim.x + threadIdx.x) >> 5);
    const int lane = threadIdx.x & 31;
    if (node >= N) return;

    const int start = row_ptr[node];
    const int end   = row_ptr[node + 1];

    float4 acc = make_float4(0.f, 0.f, 0.f, 0.f);

    for (int base = start; base < end; base += 32) {
        const int e   = base + lane;
        const int c   = (e < end) ? col_idx[e] : 0;
        const float v = (e < end) ? values[e] : 0.f;
        const int cnt = min(32, end - base);
        for (int j = 0; j < cnt; ++j) {
            const int   cj = __shfl(c, j);
            const float vj = __shfl(v, j);
            const float4 x = ((const float4*)(XW + (size_t)cj * F_DIM))[lane];
            acc.x += vj * x.x;
            acc.y += vj * x.y;
            acc.z += vj * x.z;
            acc.w += vj * x.w;
        }
    }
    ((float4*)(out + (size_t)node * F_DIM))[lane] = acc;
}

// ---------------------------------------------------------------------------
extern "C" void kernel_launch(void* const* d_in, const int* in_sizes, int n_in,
                              void* d_out, int out_size, void* d_ws, size_t ws_size,
                              hipStream_t stream) {
    const int*   row_ptr = (const int*)d_in[0];
    const int*   col_idx = (const int*)d_in[1];
    const float* values  = (const float*)d_in[2];
    const float* X       = (const float*)d_in[3];
    const float* W       = (const float*)d_in[4];
    // d_in[5] num_neighbors, d_in[6] actual_F: unused (actual_F == F_IN == 128)

    const int N = in_sizes[0] - 1;               // 100000
    float* out = (float*)d_out;

    // workspace layout: [ XW : N*128 f32 ][ Wt : 128*128 f16 ]
    float*     XW = (float*)d_ws;
    _Float16*  Wt = (_Float16*)((char*)d_ws + (size_t)N * F_DIM * sizeof(float));

    prep_w_kernel<<<1, 256, 0, stream>>>(W, Wt);

    const int numTiles = (N + 15) / 16;          // 6250, 8 tiles (waves) per block
    gemm_wmma_kernel<<<(numTiles + 7) / 8, 256, 0, stream>>>(X, Wt, XW, N);

    agg_kernel<<<(N + 7) / 8, 256, 0, stream>>>(row_ptr, col_idx, values, XW, out, N);
}
